// SelfAttention_85676007621140
// MI455X (gfx1250) — compile-verified, matrix operand order checked by reading, not run on hardware
//
#include <hip/hip_runtime.h>

typedef __bf16 bf16_t;
typedef __attribute__((ext_vector_type(16))) __bf16 v16bf;
typedef __attribute__((ext_vector_type(8)))  float  v8f;
typedef __attribute__((ext_vector_type(4)))  int    v4i_t;
typedef __attribute__((address_space(1))) v4i_t gv4i_t;   // global int4
typedef __attribute__((address_space(3))) v4i_t sv4i_t;   // LDS int4

union FragU {
    unsigned int u[8];
    v16bf v;
};

// ---------------------------------------------------------------------------
// CDNA5 async global->LDS 16-byte copy (ASYNCcnt-tracked, no VGPR round trip)
// ---------------------------------------------------------------------------
__device__ __forceinline__ void async_copy16(void* lds, const void* g) {
#if __has_builtin(__builtin_amdgcn_global_load_async_to_lds_b128)
    __builtin_amdgcn_global_load_async_to_lds_b128(
        (gv4i_t*)(uintptr_t)g,
        (sv4i_t*)lds,
        0, 0);
#else
    asm volatile("global_load_async_to_lds_b128 %0, %1, off"
                 :: "v"((unsigned int)(uintptr_t)lds), "v"(g)
                 : "memory");
#endif
}

__device__ __forceinline__ void wait_async0() {
#if __has_builtin(__builtin_amdgcn_s_wait_asynccnt)
    __builtin_amdgcn_s_wait_asynccnt(0);
#else
    asm volatile("s_wait_asynccnt 0x0" ::: "memory");
#endif
}

// ---------------------------------------------------------------------------
// fp32 -> bf16 conversion, 4 elements/thread (16B in, 8B out)
// ---------------------------------------------------------------------------
__global__ __launch_bounds__(256)
void f32_to_bf16x4_kernel(const float* __restrict__ in, bf16_t* __restrict__ out, int n4) {
    int i = blockIdx.x * blockDim.x + threadIdx.x;
    if (i < n4) {
        float4 f = ((const float4*)in)[i];
        union { bf16_t h[4]; uint2 u; } o;
        o.h[0] = (bf16_t)f.x; o.h[1] = (bf16_t)f.y;
        o.h[2] = (bf16_t)f.z; o.h[3] = (bf16_t)f.w;
        ((uint2*)out)[i] = o.u;
    }
}

// ---------------------------------------------------------------------------
// Generic bf16 WMMA GEMM with async-LDS double buffering.
//   C[M,N] = scale * A[M,K] * op(B)
//   BT = true  : op(B) = B^T, B row-major [N,K]  (projections, Q*K^T)
//   BT = false : op(B) = B,   B row-major [K,N]  (P*V)
// Block = 256 threads = 8 waves; block tile 128x128; wave tile 32x64 (4x2
// wave grid); K step 32 -> 8 v_wmma per wave per step, one dscnt wait.
// M%128==0, N%128==0, K%32==0 (true for all shapes here).
// ---------------------------------------------------------------------------
template <bool BT, typename CT>
__global__ __launch_bounds__(256)
void gemm_bf16_wmma(const bf16_t* __restrict__ A, int lda,
                    const bf16_t* __restrict__ B, int ldb,
                    CT* __restrict__ C, int ldc,
                    int M, int N, int K, float scale) {
    // Double-buffered tiles, 8KB each:
    //   A: 128 rows x 32 bf16 (row = 16 dwords)
    //   B (BT):  128 n-rows x 32 bf16   B (!BT): 32 k-rows x 128 bf16
    __shared__ __attribute__((aligned(16))) unsigned int As[2][128 * 16];
    __shared__ __attribute__((aligned(16))) unsigned int Bs[2][128 * 16];

    const int tid  = threadIdx.x;
    const int wave = tid >> 5;        // 0..7 (wave32)
    const int lane = tid & 31;
    const int l16  = lane & 15;
    const int hi   = (lane >> 4) & 1;

    const int tileM = blockIdx.y * 128;
    const int tileN = blockIdx.x * 128;
    const int mbase = (wave & 3) * 32;   // wave tile: 32 rows
    const int nbase = (wave >> 2) * 64;  //            64 cols

    v8f acc[2][4] = {};

    // ---- persistent per-thread staging state (advanced each stage call) ----
    // A (and B when BT): idx -> row r = idx>>2 (0..127), quad q = idx&3.
    // B when !BT:        idx -> krow  = idx>>4 (0..31),  quad q = idx&15.
    const bf16_t* pA[2];
    const bf16_t* pB[2];
    unsigned int ldsA[2], ldsB[2];
    #pragma unroll
    for (int i = 0; i < 2; i++) {
        int idx = tid * 2 + i;            // 0..511
        int rA  = idx >> 2, qA = idx & 3;
        pA[i]   = A + (size_t)(tileM + rA) * lda + qA * 8;
        ldsA[i] = rA * 16 + qA * 4;       // dwords
        if (BT) {
            pB[i]   = B + (size_t)(tileN + rA) * ldb + qA * 8;
            ldsB[i] = rA * 16 + qA * 4;
        } else {
            int kr = idx >> 4, qB = idx & 15;
            pB[i]   = B + (size_t)kr * ldb + tileN + qB * 8;
            ldsB[i] = kr * 64 + qB * 4;   // row = 128 bf16 = 64 dwords
        }
    }
    const size_t stepB = BT ? 32 : (size_t)32 * ldb;

    auto stage = [&](int buf) {
        #pragma unroll
        for (int i = 0; i < 2; i++) {
            async_copy16(&As[buf][ldsA[i]], pA[i]);
            pA[i] += 32;
        }
        #pragma unroll
        for (int i = 0; i < 2; i++) {
            async_copy16(&Bs[buf][ldsB[i]], pB[i]);
            pB[i] += stepB;
        }
    };

    // Prologue: DMA tile 0, wait, barrier.
    stage(0);
    wait_async0();
    __syncthreads();

    int cur = 0;
    for (int k0 = 0; k0 < K; k0 += 32) {
        const int nxt = cur ^ 1;
        if (k0 + 32 < K) stage(nxt);      // overlap DMA with WMMA below

        // ---- load ALL fragments first (single dscnt drain), then 8 WMMAs ----
        FragU a[2];
        #pragma unroll
        for (int mi = 0; mi < 2; mi++) {
            const int m = mbase + mi * 16 + l16;
            #pragma unroll
            for (int v = 0; v < 8; v++) {
                int p = v + (v >= 4 ? 4 : 0) + hi * 4;   // K-pair dword index
                a[mi].u[v] = As[cur][m * 16 + p];
            }
        }
        FragU b[4];
        #pragma unroll
        for (int nb = 0; nb < 4; nb++) {
            if (BT) {
                const int n = nbase + nb * 16 + l16;
                #pragma unroll
                for (int v = 0; v < 8; v++) {
                    int p = v + (v >= 4 ? 4 : 0) + hi * 4;
                    b[nb].u[v] = Bs[cur][n * 16 + p];
                }
            } else {
                const unsigned short* Bk = (const unsigned short*)Bs[cur];
                const int c = nbase + nb * 16 + l16;
                #pragma unroll
                for (int v = 0; v < 8; v++) {
                    int p = v + (v >= 4 ? 4 : 0) + hi * 4;
                    unsigned int lo = Bk[(2 * p + 0) * 128 + c];
                    unsigned int h2 = Bk[(2 * p + 1) * 128 + c];
                    b[nb].u[v] = lo | (h2 << 16);
                }
            }
        }

        #pragma unroll
        for (int mi = 0; mi < 2; mi++)
            #pragma unroll
            for (int nb = 0; nb < 4; nb++)
                acc[mi][nb] = __builtin_amdgcn_wmma_f32_16x16x32_bf16(
                    false, a[mi].v, false, b[nb].v, (short)0, acc[mi][nb], false, false);

        // All waves: DMA for nxt complete + everyone done reading cur.
        wait_async0();
        __syncthreads();
        cur = nxt;
    }

    // ---- store: VGPR i -> row i + 8*hi, lane%16 -> col ----
    #pragma unroll
    for (int mi = 0; mi < 2; mi++) {
        #pragma unroll
        for (int nb = 0; nb < 4; nb++) {
            #pragma unroll
            for (int i = 0; i < 8; i++) {
                int row = tileM + mbase + mi * 16 + hi * 8 + i;
                int col = tileN + nbase + nb * 16 + l16;
                C[(size_t)row * ldc + col] = (CT)(acc[mi][nb][i] * scale);
            }
        }
    }
}

// ---------------------------------------------------------------------------
// Row softmax: fp32 scores [rows, n] -> bf16 probabilities. One block per row.
// ---------------------------------------------------------------------------
__global__ __launch_bounds__(256)
void softmax_kernel(const float* __restrict__ S, bf16_t* __restrict__ P, int n) {
    __shared__ float red[256];
    const int row = blockIdx.x;
    const int tid = threadIdx.x;
    const float* s = S + (size_t)row * n;
    bf16_t* p = P + (size_t)row * n;

    float m = -__builtin_inff();
    for (int i = tid; i < n; i += 256) m = fmaxf(m, s[i]);
    red[tid] = m;
    __syncthreads();
    for (int st = 128; st > 0; st >>= 1) {
        if (tid < st) red[tid] = fmaxf(red[tid], red[tid + st]);
        __syncthreads();
    }
    m = red[0];
    __syncthreads();

    float sum = 0.f;
    for (int i = tid; i < n; i += 256) sum += __expf(s[i] - m);
    red[tid] = sum;
    __syncthreads();
    for (int st = 128; st > 0; st >>= 1) {
        if (tid < st) red[tid] += red[tid + st];
        __syncthreads();
    }
    const float inv = 1.0f / red[0];

    for (int i = tid; i < n; i += 256) p[i] = (bf16_t)(__expf(s[i] - m) * inv);
}

// ---------------------------------------------------------------------------
// Launch: bf16 casts -> Q/K/V projections -> per-batch scores/softmax/PV.
// Workspace (bf16 elems unless noted), ~96 MB:
//   Q[BSD] K[BSD] V[BSD] xb[BSD] Wq[DD] Wk[DD] Wv[DD] | scores f32[S*S] | P[S*S]
// ---------------------------------------------------------------------------
extern "C" void kernel_launch(void* const* d_in, const int* in_sizes, int n_in,
                              void* d_out, int out_size, void* d_ws, size_t ws_size,
                              hipStream_t stream) {
    (void)in_sizes; (void)n_in; (void)out_size; (void)ws_size;

    const float* x  = (const float*)d_in[0];
    const float* Wq = (const float*)d_in[1];
    const float* Wk = (const float*)d_in[2];
    const float* Wv = (const float*)d_in[3];
    float* out = (float*)d_out;

    const int Bn = 4, S = 2048, D = 1024;
    const size_t BSD = (size_t)Bn * S * D;   // 8388608
    const size_t DD  = (size_t)D * D;        // 1048576

    bf16_t* Qb  = (bf16_t*)d_ws;
    bf16_t* Kb  = Qb  + BSD;
    bf16_t* Vb  = Kb  + BSD;
    bf16_t* xb  = Vb  + BSD;
    bf16_t* Wqb = xb  + BSD;
    bf16_t* Wkb = Wqb + DD;
    bf16_t* Wvb = Wkb + DD;
    float*  Sc  = (float*)(Wvb + DD);            // S*S fp32 scores (per batch)
    bf16_t* Pb  = (bf16_t*)(Sc + (size_t)S * S); // S*S bf16 probs  (per batch)

    // 1) casts (all sizes divisible by 4)
    f32_to_bf16x4_kernel<<<dim3((unsigned)((BSD / 4 + 255) / 256)), 256, 0, stream>>>(x,  xb,  (int)(BSD / 4));
    f32_to_bf16x4_kernel<<<dim3((unsigned)((DD  / 4 + 255) / 256)), 256, 0, stream>>>(Wq, Wqb, (int)(DD / 4));
    f32_to_bf16x4_kernel<<<dim3((unsigned)((DD  / 4 + 255) / 256)), 256, 0, stream>>>(Wk, Wkb, (int)(DD / 4));
    f32_to_bf16x4_kernel<<<dim3((unsigned)((DD  / 4 + 255) / 256)), 256, 0, stream>>>(Wv, Wvb, (int)(DD / 4));

    // 2) projections: y = x @ W^T  (NT gemm, M = B*S)
    {
        dim3 grid(D / 128, (Bn * S) / 128);
        gemm_bf16_wmma<true, bf16_t><<<grid, 256, 0, stream>>>(xb, D, Wqb, D, Qb, D, Bn * S, D, D, 1.0f);
        gemm_bf16_wmma<true, bf16_t><<<grid, 256, 0, stream>>>(xb, D, Wkb, D, Kb, D, Bn * S, D, D, 1.0f);
        gemm_bf16_wmma<true, bf16_t><<<grid, 256, 0, stream>>>(xb, D, Wvb, D, Vb, D, Bn * S, D, D, 1.0f);
    }

    // 3) attention, batch-sequential (stream order serializes reuse of Sc/Pb)
    const float inv_sqrt_d = 0.03125f;  // 1/sqrt(1024)
    for (int b = 0; b < Bn; b++) {
        const size_t off = (size_t)b * S * D;
        gemm_bf16_wmma<true, float><<<dim3(S / 128, S / 128), 256, 0, stream>>>(
            Qb + off, D, Kb + off, D, Sc, S, S, S, D, inv_sqrt_d);
        softmax_kernel<<<dim3(S), 256, 0, stream>>>(Sc, Pb, S);
        gemm_bf16_wmma<false, float><<<dim3(D / 128, S / 128), 256, 0, stream>>>(
            Pb, S, Vb + off, D, out + off, D, S, D, S, 1.0f);
    }
}
